// DSP_79001628443232
// MI455X (gfx1250) — matrix operand order repaired and checked by reading, not compiled, not run on hardware
//
#include <hip/hip_runtime.h>

// ---------------------------------------------------------------------------
// MI455X / gfx1250 fused edge-message-passing pipeline, M=32 row tiles.
// All heavy GEMMs run on V_WMMA_F32_16X16X32_BF16 (fp32 accumulate); each
// B (weight) fragment register feeds two WMMAs (row tiles 0-15 / 16-31).
// ---------------------------------------------------------------------------

typedef __bf16 bf16_t;
typedef __attribute__((ext_vector_type(8)))  bf16_t v8bf;
typedef __attribute__((ext_vector_type(16))) bf16_t v16bf;
typedef __attribute__((ext_vector_type(8)))  float  v8f;

#define DD 128          // channel dim
#define TM 32           // edge/agent rows per block (2 WMMA row-tiles)
#define NTHREADS 256    // 8 wave32

static __device__ __forceinline__ v16bf cat16(v8bf lo, v8bf hi) {
  return __builtin_shufflevector(lo, hi, 0,1,2,3,4,5,6,7,8,9,10,11,12,13,14,15);
}

// A fragment (16x32 bf16, row-major source tile rows [r0..r0+15][ldk] in LDS).
// ISA layout: lane<16 holds K {k0..k0+7, k0+16..k0+23}; lane>=16 the +8 halves.
static __device__ __forceinline__ v16bf a_frag_lds(const bf16_t* buf, int ldk, int r0, int k0, int lane) {
  int m    = (lane & 15) + r0;
  int hsel = (lane >> 4) & 1;
  const bf16_t* p = buf + m * ldk + k0 + hsel * 8;
  v8bf lo = *(const v8bf*)p;
  v8bf hi = *(const v8bf*)(p + 16);
  return cat16(lo, hi);
}

// B fragment (32x16 bf16) for B = W^T, W row-major [128][ldk] bf16 in global.
// Lane n=lane&15 (+tile col base), K {k0+hsel*8..+7, k0+16+hsel*8..+7} (matches A).
static __device__ __forceinline__ v16bf b_frag_glb(const bf16_t* W, int ldk, int nc0, int k0, int lane) {
  const bf16_t* p = W + (size_t)(nc0 + (lane & 15)) * ldk + k0 + ((lane >> 4) & 1) * 8;
  v8bf lo = *(const v8bf*)p;
  v8bf hi = *(const v8bf*)(p + 16);
  return cat16(lo, hi);
}

static __device__ __forceinline__ v8f wmma_bf16(v16bf a, v16bf b, v8f c) {
  return __builtin_amdgcn_wmma_f32_16x16x32_bf16(false, a, false, b, (short)0, c, false, false);
}

// Scatter a 16x16 f32 accumulator tile into LDS [TM][128] at row base r0, col base nc0.
static __device__ __forceinline__ void store_tile_lds(float* ftile, int r0, int nc0, const v8f c, int lane) {
  int col   = nc0 + (lane & 15);
  int rbase = r0 + (lane >> 4) * 8;
#pragma unroll
  for (int r = 0; r < 8; ++r) ftile[(rbase + r) * DD + col] = c[r];
}

// Per-row GroupNorm over 128 channels on LDS f32 tile [32][128]; writes bf16
// into dst[row*ld_dst + col_ofs + c]. 256 threads: 8 threads/row, 16 ch/thread.
// Caller barriers before (ftile complete) and after (dst consumed).
static __device__ __forceinline__ void gn_to_bf16(
    const float* ftile, const float* gw, const float* gb,
    bf16_t* dst, int ld_dst, int col_ofs, bool do_relu,
    float* red, float* red2, float* stats, int tid) {
  int row = tid >> 3, seg = tid & 7, c0 = seg * 16;
  float s = 0.f, s2 = 0.f;
#pragma unroll
  for (int j = 0; j < 16; ++j) {
    float x = ftile[row * DD + c0 + j];
    s += x; s2 += x * x;
  }
  red[row * 8 + seg] = s;
  red2[row * 8 + seg] = s2;
  __syncthreads();
  if (tid < TM) {
    float S = 0.f, S2 = 0.f;
#pragma unroll
    for (int i = 0; i < 8; ++i) { S += red[tid * 8 + i]; S2 += red2[tid * 8 + i]; }
    float mean = S * (1.f / 128.f);
    float var  = S2 * (1.f / 128.f) - mean * mean;
    stats[tid * 2 + 0] = mean;
    stats[tid * 2 + 1] = rsqrtf(var + 1e-5f);
  }
  __syncthreads();
  float mean = stats[row * 2 + 0], rstd = stats[row * 2 + 1];
#pragma unroll
  for (int j = 0; j < 16; ++j) {
    int c = c0 + j;
    float y = (ftile[row * DD + c] - mean) * rstd * gw[c] + gb[c];
    if (do_relu) y = fmaxf(y, 0.f);
    dst[row * ld_dst + col_ofs + c] = (bf16_t)y;
  }
}

// ---------------------------------------------------------------------------
// fp32 -> bf16 convert (weights once per call; ctx table for L2-resident gathers)
// ---------------------------------------------------------------------------
__global__ void __launch_bounds__(256) cvt_bf16_kernel(const float* __restrict__ src,
                                                       bf16_t* __restrict__ dst, int n) {
  int i = blockIdx.x * 256 + threadIdx.x;
  if (i < n) dst[i] = (bf16_t)src[i];
}

// ---------------------------------------------------------------------------
// Agent pre-pass: q = relu(gn(agts @ Wq^T)) -> bf16 ; a_acc = agts @ Wagt^T
// ---------------------------------------------------------------------------
__global__ void __launch_bounds__(NTHREADS) agent_pre_kernel(
    const float* __restrict__ agts,
    const bf16_t* __restrict__ Wq, const float* __restrict__ gw_q, const float* __restrict__ gb_q,
    const bf16_t* __restrict__ Wagt,
    bf16_t* __restrict__ q_bf, float* __restrict__ a_acc, int N) {
  __shared__ bf16_t abuf[TM * DD];
  __shared__ bf16_t qtile[TM * DD];
  __shared__ float  ftile[TM * DD];
  __shared__ float  red[TM * 8], red2[TM * 8], stats[TM * 2];

  int tid = threadIdx.x, lane = tid & 31, wave = tid >> 5;
  int a0 = blockIdx.x * TM;
  if (a0 >= N) return;

  {  // stage agts tile as bf16 (16 elems/thread)
    int row = tid >> 3, c0 = (tid & 7) * 16;
    const float* src = agts + (size_t)(a0 + row) * DD + c0;
#pragma unroll
    for (int j = 0; j < 16; ++j) abuf[row * DD + c0 + j] = (bf16_t)src[j];
  }
  __syncthreads();

  {  // GEMM: q (K=128), two row tiles sharing each B fragment
    v8f acc0 = {}, acc1 = {};
    int nc0 = wave * 16;
#pragma unroll
    for (int k0 = 0; k0 < DD; k0 += 32) {
      v16bf b = b_frag_glb(Wq, DD, nc0, k0, lane);
      acc0 = wmma_bf16(a_frag_lds(abuf, DD, 0,  k0, lane), b, acc0);
      acc1 = wmma_bf16(a_frag_lds(abuf, DD, 16, k0, lane), b, acc1);
    }
    store_tile_lds(ftile, 0,  nc0, acc0, lane);
    store_tile_lds(ftile, 16, nc0, acc1, lane);
  }
  __syncthreads();
  gn_to_bf16(ftile, gw_q, gb_q, qtile, DD, 0, true, red, red2, stats, tid);
  __syncthreads();
  {  // q tile -> global bf16 (2 x 16B per thread)
    int row = tid >> 3, c0 = (tid & 7) * 16;
    *(v8bf*)(q_bf + (size_t)(a0 + row) * DD + c0)     = *(const v8bf*)&qtile[row * DD + c0];
    *(v8bf*)(q_bf + (size_t)(a0 + row) * DD + c0 + 8) = *(const v8bf*)&qtile[row * DD + c0 + 8];
  }
  {  // GEMM: a_acc (fp32, direct from accumulator regs)
    v8f acc0 = {}, acc1 = {};
    int nc0 = wave * 16;
#pragma unroll
    for (int k0 = 0; k0 < DD; k0 += 32) {
      v16bf b = b_frag_glb(Wagt, DD, nc0, k0, lane);
      acc0 = wmma_bf16(a_frag_lds(abuf, DD, 0,  k0, lane), b, acc0);
      acc1 = wmma_bf16(a_frag_lds(abuf, DD, 16, k0, lane), b, acc1);
    }
    int col = nc0 + (lane & 15), rbase = (lane >> 4) * 8;
#pragma unroll
    for (int r = 0; r < 8; ++r)
      a_acc[(size_t)(a0 + rbase + r) * DD + col] = acc0[r];
#pragma unroll
    for (int r = 0; r < 8; ++r)
      a_acc[(size_t)(a0 + 16 + rbase + r) * DD + col] = acc1[r];
  }
}

// ---------------------------------------------------------------------------
// Edge kernel: dist MLP -> GEMM(Wd2)+GN -> concat(d,q[hi],ctx[wi]) ->
//              GEMM(Wc1,K=384)+GN -> GEMM(Wc2) -> atomic scatter into a_acc
// ---------------------------------------------------------------------------
__global__ void __launch_bounds__(NTHREADS) edge_kernel(
    const float* __restrict__ agt_ctrs, const float* __restrict__ ctx_ctrs,
    const int* __restrict__ hi, const int* __restrict__ wi,
    const float* __restrict__ W_d1, const float* __restrict__ b_d1,
    const bf16_t* __restrict__ Wd2, const float* __restrict__ gw_d2, const float* __restrict__ gb_d2,
    const bf16_t* __restrict__ q_bf, const bf16_t* __restrict__ ctx_bf,
    const bf16_t* __restrict__ Wc1, const float* __restrict__ gw_c1, const float* __restrict__ gb_c1,
    const bf16_t* __restrict__ Wc2,
    float* __restrict__ a_acc, int E) {
  __shared__ bf16_t abuf[TM * DD];        // d1 (A of GEMM1)           8 KB
  __shared__ bf16_t cbuf[TM * 3 * DD];    // concat [d|q|ctx] (GEMM2) 24 KB
  __shared__ bf16_t ebuf[TM * DD];        // A of GEMM3                8 KB
  __shared__ float  ftile[TM * DD];       //                          16 KB
  __shared__ float  red[TM * 8], red2[TM * 8], stats[TM * 2];
  __shared__ int    hiL[TM], wiL[TM];
  __shared__ float  relL[TM * 2];

  int tid = threadIdx.x, lane = tid & 31, wave = tid >> 5;
  int e0 = blockIdx.x * TM;
  if (e0 >= E) return;

  if (tid < TM) {
    int e = e0 + tid;
    int h = hi[e], w = wi[e];
    hiL[tid] = h;
    wiL[tid] = w;
    relL[tid * 2 + 0] = agt_ctrs[h * 2 + 0] - ctx_ctrs[w * 2 + 0];
    relL[tid * 2 + 1] = agt_ctrs[h * 2 + 1] - ctx_ctrs[w * 2 + 1];
  }
  __syncthreads();

  {  // dist MLP layer 1: 2 -> 128, ReLU (pure VALU, trivially cheap)
    int row = tid >> 3, c0 = (tid & 7) * 16;
    float rx = relL[row * 2 + 0], ry = relL[row * 2 + 1];
#pragma unroll
    for (int j = 0; j < 16; ++j) {
      int c = c0 + j;
      float v = W_d1[c * 2 + 0] * rx + W_d1[c * 2 + 1] * ry + b_d1[c];
      abuf[row * DD + c] = (bf16_t)fmaxf(v, 0.f);
    }
  }
  {  // gather q[hi], ctx[wi] (bf16, L2-resident tables) into concat cols 128..383
    int row = tid >> 3, c0 = (tid & 7) * 16;
    const bf16_t* qrow = q_bf   + (size_t)hiL[row] * DD + c0;
    const bf16_t* xrow = ctx_bf + (size_t)wiL[row] * DD + c0;
    bf16_t* qd = &cbuf[row * 3 * DD + DD     + c0];
    bf16_t* xd = &cbuf[row * 3 * DD + 2 * DD + c0];
    *(v8bf*)qd       = *(const v8bf*)qrow;
    *(v8bf*)(qd + 8) = *(const v8bf*)(qrow + 8);
    *(v8bf*)xd       = *(const v8bf*)xrow;
    *(v8bf*)(xd + 8) = *(const v8bf*)(xrow + 8);
  }
  __syncthreads();

  {  // GEMM1: d1 @ Wd2^T  (K=128)
    v8f acc0 = {}, acc1 = {};
    int nc0 = wave * 16;
#pragma unroll
    for (int k0 = 0; k0 < DD; k0 += 32) {
      v16bf b = b_frag_glb(Wd2, DD, nc0, k0, lane);
      acc0 = wmma_bf16(a_frag_lds(abuf, DD, 0,  k0, lane), b, acc0);
      acc1 = wmma_bf16(a_frag_lds(abuf, DD, 16, k0, lane), b, acc1);
    }
    store_tile_lds(ftile, 0,  nc0, acc0, lane);
    store_tile_lds(ftile, 16, nc0, acc1, lane);
  }
  __syncthreads();
  gn_to_bf16(ftile, gw_d2, gb_d2, cbuf, 3 * DD, 0, true, red, red2, stats, tid);
  __syncthreads();

  {  // GEMM2: concat @ Wc1^T  (K=384) — the dominant GEMM
    v8f acc0 = {}, acc1 = {};
    int nc0 = wave * 16;
#pragma unroll
    for (int k0 = 0; k0 < 3 * DD; k0 += 32) {
      v16bf b = b_frag_glb(Wc1, 3 * DD, nc0, k0, lane);
      acc0 = wmma_bf16(a_frag_lds(cbuf, 3 * DD, 0,  k0, lane), b, acc0);
      acc1 = wmma_bf16(a_frag_lds(cbuf, 3 * DD, 16, k0, lane), b, acc1);
    }
    store_tile_lds(ftile, 0,  nc0, acc0, lane);
    store_tile_lds(ftile, 16, nc0, acc1, lane);
  }
  __syncthreads();
  gn_to_bf16(ftile, gw_c1, gb_c1, ebuf, DD, 0, true, red, red2, stats, tid);
  __syncthreads();

  {  // GEMM3: @ Wc2^T (K=128), then fp32 atomic scatter-add into a_acc[hi]
    v8f acc0 = {}, acc1 = {};
    int nc0 = wave * 16;
#pragma unroll
    for (int k0 = 0; k0 < DD; k0 += 32) {
      v16bf b = b_frag_glb(Wc2, DD, nc0, k0, lane);
      acc0 = wmma_bf16(a_frag_lds(ebuf, DD, 0,  k0, lane), b, acc0);
      acc1 = wmma_bf16(a_frag_lds(ebuf, DD, 16, k0, lane), b, acc1);
    }
    int col = nc0 + (lane & 15), rbase = (lane >> 4) * 8;
#pragma unroll
    for (int r = 0; r < 8; ++r)
      atomicAdd(&a_acc[(size_t)hiL[rbase + r] * DD + col], acc0[r]);
#pragma unroll
    for (int r = 0; r < 8; ++r)
      atomicAdd(&a_acc[(size_t)hiL[16 + rbase + r] * DD + col], acc1[r]);
  }
}

// ---------------------------------------------------------------------------
// Agent post-pass: relu(gn(a_acc)) @ Wlin^T -> gn -> +res -> relu -> out
// ---------------------------------------------------------------------------
__global__ void __launch_bounds__(NTHREADS) agent_post_kernel(
    const float* __restrict__ agts, const float* __restrict__ a_acc,
    const float* __restrict__ gw_n, const float* __restrict__ gb_n,
    const bf16_t* __restrict__ Wlin, const float* __restrict__ gw_l, const float* __restrict__ gb_l,
    float* __restrict__ out, int N) {
  __shared__ bf16_t nbuf[TM * DD];
  __shared__ float  ftile[TM * DD];
  __shared__ float  red[TM * 8], red2[TM * 8], stats[TM * 2];

  int tid = threadIdx.x, lane = tid & 31, wave = tid >> 5;
  int a0 = blockIdx.x * TM;
  if (a0 >= N) return;

  {  // stage a_acc tile
    int row = tid >> 3, c0 = (tid & 7) * 16;
    const float* src = a_acc + (size_t)(a0 + row) * DD + c0;
#pragma unroll
    for (int j = 0; j < 16; ++j) ftile[row * DD + c0 + j] = src[j];
  }
  __syncthreads();
  gn_to_bf16(ftile, gw_n, gb_n, nbuf, DD, 0, true, red, red2, stats, tid);
  __syncthreads();

  {  // GEMM: @ Wlin^T (K=128)
    v8f acc0 = {}, acc1 = {};
    int nc0 = wave * 16;
#pragma unroll
    for (int k0 = 0; k0 < DD; k0 += 32) {
      v16bf b = b_frag_glb(Wlin, DD, nc0, k0, lane);
      acc0 = wmma_bf16(a_frag_lds(nbuf, DD, 0,  k0, lane), b, acc0);
      acc1 = wmma_bf16(a_frag_lds(nbuf, DD, 16, k0, lane), b, acc1);
    }
    store_tile_lds(ftile, 0,  nc0, acc0, lane);
    store_tile_lds(ftile, 16, nc0, acc1, lane);
  }
  __syncthreads();

  {  // final GN (no relu) + residual + relu, fp32 out
    int row = tid >> 3, seg = tid & 7, c0 = seg * 16;
    float s = 0.f, s2 = 0.f;
#pragma unroll
    for (int j = 0; j < 16; ++j) {
      float x = ftile[row * DD + c0 + j];
      s += x; s2 += x * x;
    }
    red[row * 8 + seg] = s;
    red2[row * 8 + seg] = s2;
    __syncthreads();
    if (tid < TM) {
      float S = 0.f, S2 = 0.f;
#pragma unroll
      for (int i = 0; i < 8; ++i) { S += red[tid * 8 + i]; S2 += red2[tid * 8 + i]; }
      float mean = S * (1.f / 128.f);
      float var  = S2 * (1.f / 128.f) - mean * mean;
      stats[tid * 2 + 0] = mean;
      stats[tid * 2 + 1] = rsqrtf(var + 1e-5f);
    }
    __syncthreads();
    float mean = stats[row * 2 + 0], rstd = stats[row * 2 + 1];
    const float* resrow = agts + (size_t)(a0 + row) * DD;
    float* orow = out + (size_t)(a0 + row) * DD;
#pragma unroll
    for (int j = 0; j < 16; ++j) {
      int c = c0 + j;
      float y = (ftile[row * DD + c] - mean) * rstd * gw_l[c] + gb_l[c];
      orow[c] = fmaxf(y + resrow[c], 0.f);
    }
  }
}

// ---------------------------------------------------------------------------
extern "C" void kernel_launch(void* const* d_in, const int* in_sizes, int n_in,
                              void* d_out, int out_size, void* d_ws, size_t ws_size,
                              hipStream_t stream) {
  const float* agts     = (const float*)d_in[0];
  const float* agt_ctrs = (const float*)d_in[1];
  const float* ctx      = (const float*)d_in[2];
  const float* ctx_ctrs = (const float*)d_in[3];
  const int*   hi       = (const int*)d_in[4];
  const int*   wi       = (const int*)d_in[5];
  const float* W_d1  = (const float*)d_in[6];
  const float* b_d1  = (const float*)d_in[7];
  const float* W_d2  = (const float*)d_in[8];
  const float* gw_d2 = (const float*)d_in[9];
  const float* gb_d2 = (const float*)d_in[10];
  const float* W_q   = (const float*)d_in[11];
  const float* gw_q  = (const float*)d_in[12];
  const float* gb_q  = (const float*)d_in[13];
  const float* W_c1  = (const float*)d_in[14];
  const float* gw_c1 = (const float*)d_in[15];
  const float* gb_c1 = (const float*)d_in[16];
  const float* W_c2  = (const float*)d_in[17];
  const float* W_agt = (const float*)d_in[18];
  const float* gw_n  = (const float*)d_in[19];
  const float* gb_n  = (const float*)d_in[20];
  const float* W_lin = (const float*)d_in[21];
  const float* gw_l  = (const float*)d_in[22];
  const float* gb_l  = (const float*)d_in[23];

  const int N  = in_sizes[0] / DD;   // 65536 agents
  const int NC = in_sizes[2] / DD;   // 65536 ctx
  const int E  = in_sizes[4];        // 524288 edges

  // Workspace layout
  char* ws = (char*)d_ws;
  size_t o = 0;
  bf16_t* q_bf   = (bf16_t*)(ws + o); o += (size_t)N  * DD * sizeof(bf16_t);
  bf16_t* ctx_bf = (bf16_t*)(ws + o); o += (size_t)NC * DD * sizeof(bf16_t);
  float*  a_acc  = (float*) (ws + o); o += (size_t)N  * DD * sizeof(float);
  bf16_t* Wd2b  = (bf16_t*)(ws + o); o += (size_t)DD * DD * sizeof(bf16_t);
  bf16_t* Wqb   = (bf16_t*)(ws + o); o += (size_t)DD * DD * sizeof(bf16_t);
  bf16_t* Wc1b  = (bf16_t*)(ws + o); o += (size_t)DD * 3 * DD * sizeof(bf16_t);
  bf16_t* Wc2b  = (bf16_t*)(ws + o); o += (size_t)DD * DD * sizeof(bf16_t);
  bf16_t* Wagtb = (bf16_t*)(ws + o); o += (size_t)DD * DD * sizeof(bf16_t);
  bf16_t* Wlinb = (bf16_t*)(ws + o); o += (size_t)DD * DD * sizeof(bf16_t);
  (void)ws_size; (void)n_in; (void)out_size;

  auto cvt = [&](const float* s, bf16_t* d, int n) {
    cvt_bf16_kernel<<<(n + 255) / 256, 256, 0, stream>>>(s, d, n);
  };
  cvt(ctx,   ctx_bf, NC * DD);
  cvt(W_d2,  Wd2b,  DD * DD);
  cvt(W_q,   Wqb,   DD * DD);
  cvt(W_c1,  Wc1b,  DD * 3 * DD);
  cvt(W_c2,  Wc2b,  DD * DD);
  cvt(W_agt, Wagtb, DD * DD);
  cvt(W_lin, Wlinb, DD * DD);

  agent_pre_kernel<<<N / TM, NTHREADS, 0, stream>>>(
      agts, Wqb, gw_q, gb_q, Wagtb, q_bf, a_acc, N);

  edge_kernel<<<E / TM, NTHREADS, 0, stream>>>(
      agt_ctrs, ctx_ctrs, hi, wi,
      W_d1, b_d1, Wd2b, gw_d2, gb_d2,
      q_bf, ctx_bf, Wc1b, gw_c1, gb_c1, Wc2b,
      a_acc, E);

  agent_post_kernel<<<N / TM, NTHREADS, 0, stream>>>(
      agts, a_acc, gw_n, gb_n, Wlinb, gw_l, gb_l, (float*)d_out, N);
}
